// GNFConverter_29978871726464
// MI455X (gfx1250) — matrix operand order repaired and checked by reading, not compiled, not run on hardware
//
#include <hip/hip_runtime.h>
#include <hip/hip_bf16.h>

typedef __attribute__((ext_vector_type(16))) _Float16 v16h;
typedef __attribute__((ext_vector_type(8)))  float    v8f;

#define NB 16
#define NA 128
#define NP 4096
#define NT 5
#define TILE_P 16
#define WAVES 8
#define BLOCK_P (WAVES * TILE_P)          // 128 points per block
#define BLOCKS_PER_B (NP / BLOCK_P)       // 32

__device__ __forceinline__ unsigned pk2h(_Float16 a, _Float16 b) {
    union { _Float16 h[2]; unsigned u; } x;
    x.h[0] = a; x.h[1] = b;
    return x.u;
}

__device__ __forceinline__ v16h make_v16h(uint4 lo, uint4 hi) {
    union { uint4 q[2]; v16h h; } u;
    u.q[0] = lo; u.q[1] = hi;
    return u.h;
}

// raw ~1-ulp hardware ops (avoid correctly-rounded libm expansions)
__device__ __forceinline__ float sqrt_raw(float x) { return __builtin_amdgcn_sqrtf(x); }
__device__ __forceinline__ float exp2_raw(float x) { return __builtin_amdgcn_exp2f(x); }
__device__ __forceinline__ float rcp_raw(float x)  { return __builtin_amdgcn_rcpf(x); }

__launch_bounds__(256)
__global__ void gnf_kernel(const float* __restrict__ coords,
                           const int*   __restrict__ atype,
                           const float* __restrict__ query,
                           float*       __restrict__ out)
{
    // ---- LDS ----
    __shared__ __attribute__((aligned(16))) float    s_coord[NA * 3];
    __shared__ float    s_cn[NA];
    __shared__ int      s_ty[NA];
    // GEMM-1 B-fragments, all 32 lanes (upper half-wave = zeros): [cc(8)][lane(32)][8 dwords]
    __shared__ __attribute__((aligned(16))) unsigned s_b1[8 * 32 * 8];
    // GEMM-2 B-fragments of V matrix: [kc(4)][tile(3)][lane(32)][12 dwords (8 used)]
    __shared__ __attribute__((aligned(16))) unsigned s_b2[4 * 3 * 32 * 12];
    // per-wave e-weight staging (A-layout source), padded row stride 40 halves
    __shared__ __attribute__((aligned(16))) _Float16 s_e[WAVES][TILE_P][40];
    __shared__ float    s_qn[WAVES][TILE_P];
    __shared__ float    s_q [WAVES][TILE_P][4];
    __shared__ float    s_S [WAVES][TILE_P][36];

    const int tid  = threadIdx.x;
    const int lane = tid & 31;
    const int wv   = tid >> 5;
    const int hi   = lane >> 4;   // half-wave
    const int ln   = lane & 15;
    const int b    = blockIdx.x / BLOCKS_PER_B;
    const int pblk = blockIdx.x % BLOCKS_PER_B;

    // ---- block setup: coords, |c|^2, types ----
    for (int i = tid; i < NA; i += 256) {
        float x = coords[(b * NA + i) * 3 + 0];
        float y = coords[(b * NA + i) * 3 + 1];
        float z = coords[(b * NA + i) * 3 + 2];
        s_coord[i * 3 + 0] = x;
        s_coord[i * 3 + 1] = y;
        s_coord[i * 3 + 2] = z;
        s_cn[i] = x * x + y * y + z * z;
        s_ty[i] = atype[b * NA + i];
    }
    __syncthreads();

    // ---- build GEMM-1 B-fragments: K slots 0-5 = c_hi (x2), 6-8 = c_lo ----
    // item = cc*32 + lq ; lq<16 -> fragment for atom cc*16+lq ; lq>=16 -> zeros
    {
        int it = tid;                       // exactly 256 items
        int lq = it & 31;
        unsigned* dst = &s_b1[it * 8];
        if (lq < 16) {
            int a = (it >> 5) * 16 + lq;
            float cx = s_coord[a * 3 + 0];
            float cy = s_coord[a * 3 + 1];
            float cz = s_coord[a * 3 + 2];
            _Float16 chx = (_Float16)cx, chy = (_Float16)cy, chz = (_Float16)cz;
            _Float16 clx = (_Float16)(cx - (float)chx);
            _Float16 cly = (_Float16)(cy - (float)chy);
            _Float16 clz = (_Float16)(cz - (float)chz);
            dst[0] = pk2h(chx, chy);
            dst[1] = pk2h(chz, chx);
            dst[2] = pk2h(chy, chz);
            dst[3] = pk2h(clx, cly);
            dst[4] = pk2h(clz, (_Float16)0.0f);
            dst[5] = 0u; dst[6] = 0u; dst[7] = 0u;
        } else {
#pragma unroll
            for (int d = 0; d < 8; ++d) dst[d] = 0u;
        }
    }

    // ---- build GEMM-2 B-fragments of V: col = 7*t + j ; j:0-2 c_hi, 3-5 c_lo, 6 ones ----
    for (int it = tid; it < 4 * 3 * 32; it += 256) {
        int kc  = it / 96;
        int rem = it % 96;
        int tl  = rem / 32;
        int lq  = rem % 32;
        int hq  = lq >> 4;
        int col = tl * 16 + (lq & 15);
        int t   = col / 7;
        int j   = col % 7;
        unsigned* dst = &s_b2[it * 12];
#pragma unroll
        for (int v = 0; v < 8; ++v) {
            _Float16 val[2];
#pragma unroll
            for (int u = 0; u < 2; ++u) {
                int k = kc * 32 + ((2 * v + u) & 8) * 2 + hq * 8 + ((2 * v + u) & 7);   // atom index (K of GEMM-2)
                _Float16 r = (_Float16)0.0f;
                if (col < 35 && s_ty[k] == t) {
                    if (j == 6) {
                        r = (_Float16)1.0f;
                    } else {
                        float c = s_coord[k * 3 + (j % 3)];
                        _Float16 chi = (_Float16)c;
                        r = (j < 3) ? chi : (_Float16)(c - (float)chi);
                    }
                }
                val[u] = r;
            }
            dst[v] = pk2h(val[0], val[1]);
        }
    }
    __syncthreads();

    // ---- per-wave: one 16-point tile ----
    const int p0 = pblk * BLOCK_P + wv * TILE_P;

    const float* qp = &query[((size_t)b * NP + p0 + ln) * 3];
    float qx = qp[0], qy = qp[1], qz = qp[2];
    float qn = qx * qx + qy * qy + qz * qz;
    if (lane < 16) {
        s_qn[wv][ln]   = qn;
        s_q [wv][ln][0] = qx;
        s_q [wv][ln][1] = qy;
        s_q [wv][ln][2] = qz;
    }

    // A-fragment of GEMM-1: K slots 0-2 q_hi, 3-5 q_lo, 6-8 q_hi (split trick)
    _Float16 qhx = (_Float16)qx, qhy = (_Float16)qy, qhz = (_Float16)qz;
    _Float16 qlx = (_Float16)(qx - (float)qhx);
    _Float16 qly = (_Float16)(qy - (float)qhy);
    _Float16 qlz = (_Float16)(qz - (float)qhz);
    v16h a1;
#pragma unroll
    for (int k = 0; k < 16; ++k) a1[k] = (_Float16)0.0f;
    if (hi == 0) {   // lanes 0-15 hold K=0..15 (K9..15 zero)
        a1[0] = qhx; a1[1] = qhy; a1[2] = qhz;
        a1[3] = qlx; a1[4] = qly; a1[5] = qlz;
        a1[6] = qhx; a1[7] = qhy;
        a1[8] = qhz;
    } else {         // lanes 16-31 hold K=16..31 (all zero)
    }

    float qnrow[8];
#pragma unroll
    for (int v = 0; v < 8; ++v) qnrow[v] = s_qn[wv][v + 8 * hi];

    v8f acc0 = {0.f,0.f,0.f,0.f,0.f,0.f,0.f,0.f};
    v8f acc1 = acc0, acc2 = acc0;

    for (int kc = 0; kc < 4; ++kc) {
#pragma unroll
        for (int sub = 0; sub < 2; ++sub) {
            int cc   = kc * 2 + sub;
            int aidx = cc * 16 + ln;                 // this lane's atom (N of GEMM-1)
            float cn = s_cn[aidx];

            // unconditional fragment load; upper half-wave rows hold zeros
            const uint4* bp = (const uint4*)&s_b1[(cc * 32 + lane) * 8];
            v16h b1 = make_v16h(bp[0], bp[1]);

            v8f dotv = {0.f,0.f,0.f,0.f,0.f,0.f,0.f,0.f};
            dotv = __builtin_amdgcn_wmma_f32_16x16x32_f16(
                false, a1, false, b1, (short)0, dotv, false, false);

            // D layout: reg v -> row p = v+8*hi ; lane -> atom cc*16 + ln
#pragma unroll
            for (int v = 0; v < 8; ++v) {
                float d2   = qnrow[v] + cn - 2.0f * dotv[v];
                d2         = fmaxf(d2, 0.0f);
                float dist = sqrt_raw(d2);       // v_sqrt_f32 (raw)
                // scaled weight: 2^14 * exp(-dist); cancels in S/Z
                float e = exp2_raw(14.0f - dist * 1.4426950408889634f);
                s_e[wv][v + 8 * hi][sub * 16 + ln] = (_Float16)e;
            }
        }

        // A-fragment of GEMM-2 from staged e-weights (row = point = ln)
        const uint4* e0 = (const uint4*)&s_e[wv][ln][hi * 8];
        const uint4* e1 = (const uint4*)&s_e[wv][ln][16 + hi * 8];
        v16h a2 = make_v16h(*e0, *e1);

        const uint4* b20 = (const uint4*)&s_b2[((kc * 3 + 0) * 32 + lane) * 12];
        const uint4* b21 = (const uint4*)&s_b2[((kc * 3 + 1) * 32 + lane) * 12];
        const uint4* b22 = (const uint4*)&s_b2[((kc * 3 + 2) * 32 + lane) * 12];
        v16h f0 = make_v16h(b20[0], b20[1]);
        v16h f1 = make_v16h(b21[0], b21[1]);
        v16h f2 = make_v16h(b22[0], b22[1]);

        acc0 = __builtin_amdgcn_wmma_f32_16x16x32_f16(false, a2, false, f0, (short)0, acc0, false, false);
        acc1 = __builtin_amdgcn_wmma_f32_16x16x32_f16(false, a2, false, f1, (short)0, acc1, false, false);
        acc2 = __builtin_amdgcn_wmma_f32_16x16x32_f16(false, a2, false, f2, (short)0, acc2, false, false);
    }

    // ---- stage S[p][col] (cols 0..34 used) ----
#pragma unroll
    for (int v = 0; v < 8; ++v) {
        int p = v + 8 * hi;
        s_S[wv][p][ln]      = acc0[v];
        s_S[wv][p][16 + ln] = acc1[v];
        if (ln < 4) s_S[wv][p][32 + ln] = acc2[v];
    }

    // ---- finalize: grad = S/Z - q, clip, store ----
#pragma unroll
    for (int i = 0; i < 3; ++i) {
        int idx = i * 32 + lane;
        if (idx < TILE_P * NT) {
            int p = idx / NT;
            int t = idx % NT;
            float Z  = s_S[wv][p][t * 7 + 6];
            float gx = 0.0f, gy = 0.0f, gz = 0.0f;
            if (Z > 0.0f) {
                float inv = rcp_raw(Z);
                gx = (s_S[wv][p][t * 7 + 0] + s_S[wv][p][t * 7 + 3]) * inv - s_q[wv][p][0];
                gy = (s_S[wv][p][t * 7 + 1] + s_S[wv][p][t * 7 + 4]) * inv - s_q[wv][p][1];
                gz = (s_S[wv][p][t * 7 + 2] + s_S[wv][p][t * 7 + 5]) * inv - s_q[wv][p][2];
                float mag = sqrt_raw(gx * gx + gy * gy + gz * gz);
                float f   = fminf(1.0f, 0.3f * rcp_raw(fmaxf(mag, 1e-12f)));
                gx *= f; gy *= f; gz *= f;
            }
            size_t o = (((size_t)b * NP + p0 + p) * NT + t) * 3;
            out[o + 0] = gx;
            out[o + 1] = gy;
            out[o + 2] = gz;
        }
    }
}

extern "C" void kernel_launch(void* const* d_in, const int* in_sizes, int n_in,
                              void* d_out, int out_size, void* d_ws, size_t ws_size,
                              hipStream_t stream) {
    (void)in_sizes; (void)n_in; (void)out_size; (void)d_ws; (void)ws_size;
    const float* coords = (const float*)d_in[0];
    const int*   atype  = (const int*)d_in[1];
    const float* query  = (const float*)d_in[2];
    float*       out    = (float*)d_out;

    dim3 grid(NB * BLOCKS_PER_B);   // 512 blocks
    dim3 block(256);                // 8 wave32
    gnf_kernel<<<grid, block, 0, stream>>>(coords, atype, query, out);
}
